// Encoder_Resnet_after_se3ACN_31954556682984
// MI455X (gfx1250) — matrix operand order repaired and checked by reading, not compiled, not run on hardware
//
#include <hip/hip_runtime.h>
#include <hip/hip_bf16.h>

// ---------------- problem constants ----------------
#define BATCH   4
#define NATOMS  286
#define BN      (BATCH * NATOMS)
#define EMB     4
#define NB      3
#define HDIM    150
#define HP      160      // HDIM padded to multiple of 32 (WMMA K)
#define CD      8
#define NCL     3
#define HID     128
#define MT      64       // j-tile rows per block iteration

typedef __attribute__((ext_vector_type(16))) _Float16 v16h;
typedef __attribute__((ext_vector_type(8)))  _Float16 v8h;
typedef __attribute__((ext_vector_type(8)))  float    v8f;

// softplus(beta=5) = log1p(exp(5x))/5, branchless & overflow-safe:
//   softplus(t) = max(t,0) + ln(1 + exp(-|t|))
// The log argument is in [1,2] (never denormal/inf), so raw v_log_f32/v_exp_f32
// (base-2 hardware transcendentals) are exact enough — no libm guard code.
static __device__ __forceinline__ float sp5(float x) {
    const float e = __builtin_amdgcn_exp2f(-7.2134752f * fabsf(x));  // exp(-|5x|)
    const float l = __builtin_amdgcn_logf(1.0f + e);                 // log2(1+e)
    return fmaxf(x, 0.0f) + 0.13862944f * l;                         // *= ln2/5
}

static __device__ __forceinline__ v8f zero8() {
    v8f c;
#pragma unroll
    for (int q = 0; q < 8; ++q) c[q] = 0.0f;
    return c;
}

// ---- WMMA fragment loaders (wave32, 16x16x32 f16) ----
// A (16x32, MxK): lanes 0-15 row M=lane, K = {0..7,16..23}; lanes 16-31 row M=lane-16, K = {8..15,24..31}
static __device__ __forceinline__ v16h ldsA(const _Float16* base, int rowbase, int stride,
                                            int kbase, int lane) {
    const int mrow = lane & 15;
    const int ksel = (lane < 16) ? 0 : 8;
    const _Float16* p = base + (rowbase + mrow) * stride + kbase;
    const v8h lo = *(const v8h*)(p + ksel);
    const v8h hi = *(const v8h*)(p + 16 + ksel);
    return __builtin_shufflevector(lo, hi, 0, 1, 2, 3, 4, 5, 6, 7,
                                   8, 9, 10, 11, 12, 13, 14, 15);
}

// B (32x16, KxN) from global weights stored transposed as [n][k] (k contiguous):
// lanes 0-15: col N=lane, K = {0..7,16..23} ; lanes 16-31: col N=lane-16, K = {8..15,24..31}
static __device__ __forceinline__ v16h glbB(const _Float16* Wt, int nt, int kstride,
                                            int kbase, int lane) {
    const int ncol = lane & 15;
    const int koff = (lane < 16) ? 0 : 8;
    const _Float16* p = Wt + (size_t)(nt * 16 + ncol) * kstride + kbase + koff;
    const v8h lo = *(const v8h*)(p);
    const v8h hi = *(const v8h*)(p + 16);
    return __builtin_shufflevector(lo, hi, 0, 1, 2, 3, 4, 5, 6, 7,
                                   8, 9, 10, 11, 12, 13, 14, 15);
}

static __device__ __forceinline__ v8f wmma16(v16h a, v16h b, v8f c) {
    return __builtin_amdgcn_wmma_f32_16x16x32_f16(false, a, false, b, (short)0, c, false, false);
}

// bias + softplus + f16 store of one 16x16 D tile (column-major epilogue)
static __device__ __forceinline__ void store_tile(v8f c, float bv, _Float16* dst,
                                                  int mt, int ncol, int lane) {
    const int rbase = mt * 16 + ((lane < 16) ? 0 : 8);
#pragma unroll
    for (int g = 0; g < 8; ++g)
        dst[(rbase + g) * HP + ncol] = (_Float16)sp5(c[g] + bv);
}

// One MLP layer over a 64x160 tile: dst = softplus5(A*W + biasPad), f16 -> LDS.
// 40 output tiles (4 row x 10 col). Wave w: full strip nt=w (B reused over 4 row
// tiles, 4 accumulators), plus 1 tile of the remainder strips nt=8,9. 5 tiles/wave.
static __device__ __forceinline__ void gemm_layer(const _Float16* Asrc, int aStride, int nkt,
                                                  const _Float16* Wt, int wStride,
                                                  const float* biasPad,
                                                  _Float16* dst, int wave, int lane) {
    const int ncl = lane & 15;
    // phase 1: full column strip nt = wave
    {
        const int nt = wave;
        v8f c0 = zero8(), c1 = zero8(), c2 = zero8(), c3 = zero8();
#pragma clang loop unroll(disable)
        for (int kt = 0; kt < nkt; ++kt) {
            const v16h bf = glbB(Wt, nt, wStride, kt * 32, lane);
            c0 = wmma16(ldsA(Asrc, 0,  aStride, kt * 32, lane), bf, c0);
            c1 = wmma16(ldsA(Asrc, 16, aStride, kt * 32, lane), bf, c1);
            c2 = wmma16(ldsA(Asrc, 32, aStride, kt * 32, lane), bf, c2);
            c3 = wmma16(ldsA(Asrc, 48, aStride, kt * 32, lane), bf, c3);
        }
        const int ncol = nt * 16 + ncl;
        const float bv = biasPad[ncol];
        store_tile(c0, bv, dst, 0, ncol, lane);
        store_tile(c1, bv, dst, 1, ncol, lane);
        store_tile(c2, bv, dst, 2, ncol, lane);
        store_tile(c3, bv, dst, 3, ncol, lane);
    }
    // phase 2: remainder strips nt = 8,9 -> one tile per wave
    {
        const int mt = wave & 3;
        const int nt = 8 + (wave >> 2);
        v8f c = zero8();
#pragma clang loop unroll(disable)
        for (int kt = 0; kt < nkt; ++kt) {
            const v16h bf = glbB(Wt, nt, wStride, kt * 32, lane);
            c = wmma16(ldsA(Asrc, mt * 16, aStride, kt * 32, lane), bf, c);
        }
        const int ncol = nt * 16 + ncl;
        store_tile(c, biasPad[ncol], dst, mt, ncol, lane);
    }
}

// ---------------- fused per-cloud kernel ----------------
// block = one (b,i) output row; loops j in tiles of 64; 256 threads (8 waves).
template <int DIN>
__global__ __launch_bounds__(256)
__attribute__((amdgpu_waves_per_eu(4)))
void cloud_kernel(
    const float* __restrict__ xyz, const float* __restrict__ featin, int finStride,
    const _Float16* __restrict__ W0t, const _Float16* __restrict__ W1t,
    const _Float16* __restrict__ W2t, const _Float16* __restrict__ Kwt,
    const float* __restrict__ b0p, const float* __restrict__ b1p,
    const float* __restrict__ b2p, float* __restrict__ featout) {
    constexpr int KOUT = 8 * DIN;

    __shared__ __align__(32) _Float16 h0s[MT * HP];   // 20 KB
    __shared__ __align__(32) _Float16 h1s[MT * HP];   // 20 KB
    __shared__ __align__(32) unsigned char scraw[MT * 64 * 4];  // 16 KB: A0 / Karr / reduction
    __shared__ float fin_s[MT * 8];
    __shared__ float mask_s[MT];

    const int tid  = threadIdx.x;
    const int lane = tid & 31;
    const int wave = tid >> 5;
    const int bi   = blockIdx.x;
    const int bb   = bi / NATOMS;

    const float xi = xyz[bi * 3 + 0];
    const float yi = xyz[bi * 3 + 1];
    const float zi = xyz[bi * 3 + 2];

    _Float16* A0 = (_Float16*)scraw;
    float* Karr  = (float*)scraw;

    float acc0 = 0.0f, acc1 = 0.0f, nbacc = 0.0f;
    const int myrow = tid & 63;
    const int og    = tid >> 6;   // o-pair group 0..3

    for (int j0 = 0; j0 < NATOMS; j0 += MT) {
        // ---- geometry + basis + gather inputs (threads 0..63, one j-row each) ----
        if (tid < MT) {
            const int j = j0 + tid;
            float m = 0.0f, bas0 = 0.0f, bas1 = 0.0f, bas2 = 0.0f;
            float fv[8];
#pragma unroll
            for (int p = 0; p < 8; ++p) fv[p] = 0.0f;
            if (j < NATOMS) {
                const float dx = xyz[(bb * NATOMS + j) * 3 + 0] - xi;
                const float dy = xyz[(bb * NATOMS + j) * 3 + 1] - yi;
                const float dz = xyz[(bb * NATOMS + j) * 3 + 2] - zi;
                const float r  = sqrtf(dx * dx + dy * dy + dz * dz + 1e-12f);
                m = (r < 3.0f) ? 1.0f : 0.0f;
                const float d0 = r * (1.0f / 1.5f);
                const float d1 = (r - 1.5f) * (1.0f / 1.5f);
                const float d2 = (r - 3.0f) * (1.0f / 1.5f);
                const float c0 = __cosf(1.57079632679f * d0);
                const float c1 = __cosf(1.57079632679f * d1);
                const float c2 = __cosf(1.57079632679f * d2);
                bas0 = (fabsf(d0) < 1.0f) ? c0 * c0 : 0.0f;
                bas1 = (fabsf(d1) < 1.0f) ? c1 * c1 : 0.0f;
                bas2 = (fabsf(d2) < 1.0f) ? c2 * c2 : 0.0f;
#pragma unroll
                for (int p = 0; p < DIN; ++p)
                    fv[p] = featin[(bb * NATOMS + j) * finStride + p];
            }
            mask_s[tid] = m;
#pragma unroll
            for (int p = 0; p < 8; ++p) fin_s[tid * 8 + p] = fv[p];
            v8h z;
#pragma unroll
            for (int q = 0; q < 8; ++q) z[q] = (_Float16)0.0f;
            v8h first = z;
            first[0] = (_Float16)bas0;
            first[1] = (_Float16)bas1;
            first[2] = (_Float16)bas2;
            _Float16* row = A0 + tid * 32;
            *(v8h*)(row + 0)  = first;
            *(v8h*)(row + 8)  = z;
            *(v8h*)(row + 16) = z;
            *(v8h*)(row + 24) = z;
        }
        __syncthreads();

        // ---- radial MLP: 3 -> 150 -> 150 -> 150 (all WMMA) ----
        gemm_layer(A0, 32, 1, W0t, 32, b0p, h0s, wave, lane);
        __syncthreads();
        gemm_layer(h0s, HP, 5, W1t, HP, b1p, h1s, wave, lane);
        __syncthreads();
        gemm_layer(h1s, HP, 5, W2t, HP, b2p, h0s, wave, lane);
        __syncthreads();

        // ---- kernel projection: [64x160] x [160 x KOUT] -> Karr f32 ----
        constexpr int NTK = 4 * (KOUT / 16);
        for (int tile = wave; tile < NTK; tile += 8) {
            const int mt = tile & 3;
            const int nt = tile >> 2;
            v8f c = zero8();
#pragma clang loop unroll(disable)
            for (int kt = 0; kt < 5; ++kt) {
                const v16h bf = glbB(Kwt, nt, HP, kt * 32, lane);
                c = wmma16(ldsA(h0s, mt * 16, HP, kt * 32, lane), bf, c);
            }
            const int ncol  = nt * 16 + (lane & 15);
            const int rbase = mt * 16 + ((lane < 16) ? 0 : 8);
#pragma unroll
            for (int g = 0; g < 8; ++g)
                Karr[(rbase + g) * KOUT + ncol] = c[g];
        }
        __syncthreads();

        // ---- masked einsum accumulation: feat_out[o] += K[j][o,p] * feat_in[j][p] ----
        {
            const float m = mask_s[myrow];
            float s0 = 0.0f, s1 = 0.0f;
#pragma unroll
            for (int p = 0; p < DIN; ++p) {
                const float fp = fin_s[myrow * 8 + p];
                s0 += Karr[myrow * KOUT + (og * 2 + 0) * DIN + p] * fp;
                s1 += Karr[myrow * KOUT + (og * 2 + 1) * DIN + p] * fp;
            }
            acc0 += s0 * m;
            acc1 += s1 * m;
            if (og == 0) nbacc += m;
        }
        __syncthreads();
    }

    // ---- reduction over the 64 row-slots, apply 1/sqrt(n_nb) ----
    float* red = (float*)scraw;
    red[tid * 2 + 0] = acc0;
    red[tid * 2 + 1] = acc1;
    float* nbr = red + 512;
    if (og == 0) nbr[myrow] = nbacc;
    __syncthreads();
    if (tid < 8) {
        const int o = tid, g = o >> 1, u = o & 1;
        float s = 0.0f, nn = 0.0f;
        for (int r = 0; r < 64; ++r) {
            s  += red[(g * 64 + r) * 2 + u];
            nn += nbr[r];
        }
        nn = fmaxf(nn, 1.0f);
        featout[bi * 24 + o] = s * rsqrtf(nn);
    }
}

// ---------------- prep kernels ----------------
__global__ void embed_kernel(const int* __restrict__ Z, const float* __restrict__ emb,
                             float* __restrict__ feat0) {
    const int idx = blockIdx.x * blockDim.x + threadIdx.x;
    if (idx >= BN * EMB) return;
    const int n = idx >> 2, p = idx & 3;
    feat0[idx] = emb[Z[n] * EMB + p];
}

// dst[n*KP + k] = (k<K && n<Nn) ? src[k*srcStride + n] : 0   (f32 -> f16, transpose+pad)
__global__ void transpose_pad(const float* __restrict__ src, _Float16* __restrict__ dst,
                              int K, int Nn, int srcStride, int KP, int NP) {
    const int idx = blockIdx.x * blockDim.x + threadIdx.x;
    if (idx >= NP * KP) return;
    const int n = idx / KP, k = idx % KP;
    const float v = (k < K && n < Nn) ? src[k * srcStride + n] : 0.0f;
    dst[idx] = (_Float16)v;
}

// padded bias: dst[c*HP + n] = (n < HDIM) ? src[c*HDIM + n] : 0, for c in [0,NCL)
__global__ void pad_bias(const float* __restrict__ src, float* __restrict__ dst) {
    const int idx = blockIdx.x * blockDim.x + threadIdx.x;
    if (idx >= NCL * HP) return;
    const int c = idx / HP, n = idx % HP;
    dst[idx] = (n < HDIM) ? src[c * HDIM + n] : 0.0f;
}

// ---------------- ResnetPointnet kernels (fp32, ~1% of FLOPs) ----------------
__global__ void pn_fcpos(const float* __restrict__ feats, const float* __restrict__ w,
                         const float* __restrict__ b, float* __restrict__ out) {
    const int idx = blockIdx.x * blockDim.x + threadIdx.x;
    if (idx >= BN * 256) return;
    const int o = idx & 255, r = idx >> 8;
    float s = b[o];
    for (int k = 0; k < 24; ++k) s += feats[r * 24 + k] * w[k * 256 + o];
    out[idx] = s;
}

__global__ void pn_stageA(const float* __restrict__ x, int xs, const float* __restrict__ pooled,
                          int Din, const float* __restrict__ w, const float* __restrict__ b,
                          float* __restrict__ t1) {
    const int idx = blockIdx.x * blockDim.x + threadIdx.x;
    if (idx >= BN * HID) return;
    const int o = idx & 127, r = idx >> 7, bb = r / NATOMS;
    float s = b[o];
    for (int k = 0; k < Din; ++k) {
        const float xv = (pooled && k >= HID) ? pooled[bb * HID + (k - HID)] : x[r * xs + k];
        s += fmaxf(xv, 0.0f) * w[k * HID + o];
    }
    t1[idx] = fmaxf(s, 0.0f);   // store relu(net1)
}

__global__ void pn_stageB(const float* __restrict__ x, int xs, const float* __restrict__ pooled,
                          int Din, const float* __restrict__ w1, const float* __restrict__ b1,
                          const float* __restrict__ sw, const float* __restrict__ t1,
                          float* __restrict__ out) {
    const int idx = blockIdx.x * blockDim.x + threadIdx.x;
    if (idx >= BN * HID) return;
    const int o = idx & 127, r = idx >> 7, bb = r / NATOMS;
    float s = b1[o];
    for (int k = 0; k < HID; ++k) s += t1[r * HID + k] * w1[k * HID + o];
    for (int k = 0; k < Din; ++k) {
        const float xv = (pooled && k >= HID) ? pooled[bb * HID + (k - HID)] : x[r * xs + k];
        s += xv * sw[k * HID + o];
    }
    out[idx] = s;
}

__global__ void pn_pool(const float* __restrict__ net, float* __restrict__ pooled) {
    const int idx = blockIdx.x * blockDim.x + threadIdx.x;
    if (idx >= BATCH * HID) return;
    const int b = idx >> 7, o = idx & 127;
    float m = -1e30f;
    for (int n = 0; n < NATOMS; ++n) m = fmaxf(m, net[(b * NATOMS + n) * HID + o]);
    pooled[idx] = m;
}

__global__ void pn_final(const float* __restrict__ net, float* __restrict__ out) {
    const int idx = blockIdx.x * blockDim.x + threadIdx.x;
    if (idx >= BATCH * HID) return;
    const int b = idx >> 7, o = idx & 127;
    float s = 0.0f;
    for (int n = 0; n < NATOMS; ++n) {
        const float v = net[(b * NATOMS + n) * HID + o];
        s += v * v;
    }
    out[idx] = sqrtf(s + 1e-12f);
}

// ---------------- host launcher ----------------
static inline int cdiv(int a, int b) { return (a + b - 1) / b; }

extern "C" void kernel_launch(void* const* d_in, const int* in_sizes, int n_in,
                              void* d_out, int out_size, void* d_ws, size_t ws_size,
                              hipStream_t stream) {
    (void)in_sizes; (void)n_in; (void)out_size; (void)ws_size;
    const float* xyz      = (const float*)d_in[0];
    const int*   Z        = (const int*)  d_in[1];
    const float* emb_w    = (const float*)d_in[2];
    const float* rad_w0   = (const float*)d_in[3];
    const float* rad_b0   = (const float*)d_in[4];
    const float* rad_w1   = (const float*)d_in[5];
    const float* rad_b1   = (const float*)d_in[6];
    const float* rad_w2   = (const float*)d_in[7];
    const float* rad_b2   = (const float*)d_in[8];
    const float* kern_w0  = (const float*)d_in[9];
    const float* kern_w12 = (const float*)d_in[10];
    const float* fc_pos_w = (const float*)d_in[11];
    const float* fc_pos_b = (const float*)d_in[12];
    const float* pn_fc0_w = (const float*)d_in[13];
    const float* pn_fc0_b = (const float*)d_in[14];
    const float* pn_fc1_w = (const float*)d_in[15];
    const float* pn_fc1_b = (const float*)d_in[16];
    const float* pn_sc_w  = (const float*)d_in[17];
    float* out = (float*)d_out;

    // workspace carve-up (256B aligned)
    unsigned char* ws = (unsigned char*)d_ws;
    size_t off = 0;
    auto carve = [&](size_t bytes) {
        unsigned char* p = ws + off;
        off = (off + bytes + 255) & ~(size_t)255;
        return p;
    };
    float*     feat0    = (float*)carve((size_t)BN * EMB * 4);
    float*     features = (float*)carve((size_t)BN * 24 * 4);
    _Float16*  W0t      = (_Float16*)carve((size_t)NCL * HP * 32 * 2);
    _Float16*  W1t      = (_Float16*)carve((size_t)NCL * HP * HP * 2);
    _Float16*  W2t      = (_Float16*)carve((size_t)NCL * HP * HP * 2);
    _Float16*  Kw0t     = (_Float16*)carve((size_t)32 * HP * 2);
    _Float16*  Kw12t    = (_Float16*)carve((size_t)2 * 64 * HP * 2);
    float*     b0p      = (float*)carve((size_t)NCL * HP * 4);
    float*     b1p      = (float*)carve((size_t)NCL * HP * 4);
    float*     b2p      = (float*)carve((size_t)NCL * HP * 4);
    float*     net0     = (float*)carve((size_t)BN * 256 * 4);
    float*     net1     = (float*)carve((size_t)BN * HID * 4);
    float*     net2     = (float*)carve((size_t)BN * HID * 4);
    float*     t1buf    = (float*)carve((size_t)BN * HID * 4);
    float*     pooled   = (float*)carve((size_t)BATCH * HID * 4);

    // ---- prep: embedding + weight conversion (transpose, pad, f16) + padded biases ----
    embed_kernel<<<cdiv(BN * EMB, 256), 256, 0, stream>>>(Z, emb_w, feat0);
    for (int c = 0; c < NCL; ++c) {
        transpose_pad<<<cdiv(HP * 32, 256), 256, 0, stream>>>(
            rad_w0 + (size_t)c * NB * HDIM, W0t + (size_t)c * HP * 32, NB, HDIM, HDIM, 32, HP);
        transpose_pad<<<cdiv(HP * HP, 256), 256, 0, stream>>>(
            rad_w1 + (size_t)c * HDIM * HDIM, W1t + (size_t)c * HP * HP, HDIM, HDIM, HDIM, HP, HP);
        transpose_pad<<<cdiv(HP * HP, 256), 256, 0, stream>>>(
            rad_w2 + (size_t)c * HDIM * HDIM, W2t + (size_t)c * HP * HP, HDIM, HDIM, HDIM, HP, HP);
    }
    transpose_pad<<<cdiv(32 * HP, 256), 256, 0, stream>>>(kern_w0, Kw0t, HDIM, 32, 32, HP, 32);
    for (int c = 0; c < 2; ++c)
        transpose_pad<<<cdiv(64 * HP, 256), 256, 0, stream>>>(
            kern_w12 + (size_t)c * HDIM * 64, Kw12t + (size_t)c * 64 * HP, HDIM, 64, 64, HP, 64);
    pad_bias<<<cdiv(NCL * HP, 256), 256, 0, stream>>>(rad_b0, b0p);
    pad_bias<<<cdiv(NCL * HP, 256), 256, 0, stream>>>(rad_b1, b1p);
    pad_bias<<<cdiv(NCL * HP, 256), 256, 0, stream>>>(rad_b2, b2p);

    // ---- clouds (sequential dependence; stream-ordered) ----
    cloud_kernel<4><<<BN, 256, 0, stream>>>(
        xyz, feat0, EMB, W0t, W1t, W2t, Kw0t,
        b0p, b1p, b2p, features + 0);
    cloud_kernel<8><<<BN, 256, 0, stream>>>(
        xyz, features + 0, 24, W0t + HP * 32, W1t + HP * HP, W2t + HP * HP, Kw12t,
        b0p + HP, b1p + HP, b2p + HP, features + 8);
    cloud_kernel<8><<<BN, 256, 0, stream>>>(
        xyz, features + 8, 24, W0t + 2 * HP * 32, W1t + 2 * HP * HP, W2t + 2 * HP * HP,
        Kw12t + 64 * HP, b0p + 2 * HP, b1p + 2 * HP, b2p + 2 * HP, features + 16);

    // ---- ResnetPointnet ----
    pn_fcpos<<<cdiv(BN * 256, 256), 256, 0, stream>>>(features, fc_pos_w, fc_pos_b, net0);

    // block 0: Din=256, input net0 (stride 256), no pooled
    pn_stageA<<<cdiv(BN * HID, 256), 256, 0, stream>>>(
        net0, 256, nullptr, 256, pn_fc0_w, pn_fc0_b, t1buf);
    pn_stageB<<<cdiv(BN * HID, 256), 256, 0, stream>>>(
        net0, 256, nullptr, 256, pn_fc1_w, pn_fc1_b, pn_sc_w, t1buf, net1);

    float* cur = net1;
    float* nxt = net2;
    for (int i = 1; i < 5; ++i) {
        const float* fc0w = pn_fc0_w + (size_t)i * 256 * HID;
        const float* fc0b = pn_fc0_b + (size_t)i * HID;
        const float* fc1w = pn_fc1_w + (size_t)i * HID * HID;
        const float* fc1b = pn_fc1_b + (size_t)i * HID;
        const float* scw  = pn_sc_w  + (size_t)i * 256 * HID;
        pn_pool<<<cdiv(BATCH * HID, 256), 256, 0, stream>>>(cur, pooled);
        pn_stageA<<<cdiv(BN * HID, 256), 256, 0, stream>>>(
            cur, HID, pooled, 256, fc0w, fc0b, t1buf);
        pn_stageB<<<cdiv(BN * HID, 256), 256, 0, stream>>>(
            cur, HID, pooled, 256, fc1w, fc1b, scw, t1buf, nxt);
        float* tmp = cur; cur = nxt; nxt = tmp;
    }

    pn_final<<<cdiv(BATCH * HID, 256), 256, 0, stream>>>(cur, out);
}
